// NMSTPP_21818433863879
// MI455X (gfx1250) — matrix operand order repaired and checked by reading, not compiled, not run on hardware
//
#include <hip/hip_runtime.h>
#include <hip/hip_bf16.h>
#include <math.h>

// ---------------------------------------------------------------------------
// Problem constants
// ---------------------------------------------------------------------------
#define NB   2048            // batch
#define NS   40              // seq len
#define NTOK (NB * NS)       // 81920 tokens
#define DMODEL 64
#define NHEAD  8
#define DHEAD  8
#define DFFN   256

typedef __attribute__((ext_vector_type(16))) _Float16 v16h;
typedef __attribute__((ext_vector_type(8)))  _Float16 v8h;
typedef __attribute__((ext_vector_type(8)))  float    v8f;

// ---------------------------------------------------------------------------
// Workspace arena layout (bytes, all offsets multiple of 256)
// ---------------------------------------------------------------------------
static constexpr size_t SZ_SRC32  = (size_t)NTOK * 64 * 4;   // 20,971,520
static constexpr size_t SZ_SRC16  = (size_t)NTOK * 64 * 2;   // 10,485,760
static constexpr size_t OFF_SRC32 = 0;
static constexpr size_t OFF_SRC16 = OFF_SRC32 + SZ_SRC32;
static constexpr size_t OFF_PK_GL0 = OFF_SRC16 + SZ_SRC16;   // 4*4*512 f16 = 16384 B
static constexpr size_t OFF_PK_GL1 = OFF_PK_GL0 + 16384;     // 4*2*512 f16 =  8192 B
static constexpr size_t OFF_PK_GL2 = OFF_PK_GL1 + 8192;      // 1*2*512 f16 =  2048 B
static constexpr size_t OFF_PK_QKV = OFF_PK_GL2 + 2048;      // 12*2*512 f16= 24576 B
static constexpr size_t OFF_ARENA  = OFF_PK_QKV + 24576;
// Stage-A scratch inside arena:
static constexpr size_t OFF_HP = OFF_ARENA;                        // NTOK*128 f16
static constexpr size_t OFF_G0 = OFF_HP + (size_t)NTOK * 128 * 2;  // NTOK*64 f16
static constexpr size_t OFF_G1 = OFF_G0 + (size_t)NTOK * 64 * 2;   // NTOK*64 f16
static constexpr size_t OFF_G2 = OFF_G1 + (size_t)NTOK * 64 * 2;   // NTOK*16 f16
static constexpr size_t SZ_ARENA = (size_t)NTOK * 128 * 2 + 2 * (size_t)NTOK * 64 * 2
                                 + (size_t)NTOK * 16 * 2;          // 44,564,480
// qkv (f16, NTOK*192*2 = 31,457,280 B) aliases the arena after stage A is done
static constexpr size_t OFF_QKV  = OFF_ARENA;
static constexpr size_t OFF_ATTN = OFF_ARENA + SZ_ARENA;           // NB*64 f32

// ---------------------------------------------------------------------------
// Weight packing into CDNA5 WMMA B-fragment lane layout.
// B (K x N) f16 32x16-per-step fragment: lane L holds col n = nt*16+(L&15),
// element i (0..15) holds K = ks*32 + 8*(L>>4) + (i&7) + 16*(i>>3). Stored contiguously so a
// GEMM lane does one 32B load per k-step:  Bp[((nt*Ks+ks)*32 + L)*16 + i]
// ---------------------------------------------------------------------------
__global__ void pack_b(const float* __restrict__ W, _Float16* __restrict__ Bp,
                       int Korig, int N, int Ksteps, int Ntiles) {
  int idx = blockIdx.x * 256 + threadIdx.x;
  int total = Ntiles * Ksteps * 512;
  if (idx >= total) return;
  int i    = idx & 15;
  int lane = (idx >> 4) & 31;
  int ks   = (idx >> 9) % Ksteps;
  int nt   = idx / (Ksteps * 512);
  int k = ks * 32 + 8 * (lane >> 4) + (i & 7) + 16 * (i >> 3);
  int n = nt * 16 + (lane & 15);
  float v = (k < Korig) ? W[(size_t)k * N + n] : 0.0f;
  Bp[idx] = (_Float16)v;
}

// ---------------------------------------------------------------------------
// Stage A1: per-token feature extraction (one wave32 per token).
// Produces hp (f16, K padded 110->128) and src[:,0:48] (+pos-enc) in f32.
// ---------------------------------------------------------------------------
__global__ __launch_bounds__(256) void stage_a1(
    const float* __restrict__ X, const float* __restrict__ emb_act,
    const float* __restrict__ emb_zone, const float* __restrict__ lin0_W,
    const float* __restrict__ lin0_b, const float* __restrict__ gat_W,
    const float* __restrict__ gat_b, const float* __restrict__ gat_a,
    _Float16* __restrict__ hp16, float* __restrict__ src32) {
  __shared__ float sx[8][118];
  __shared__ float sh[8][110];
  __shared__ float ss1[8][22];
  __shared__ float ss2[8][22];

  const int w    = threadIdx.x >> 5;
  const int lane = threadIdx.x & 31;
  const int tok  = blockIdx.x * 8 + w;   // grid = NTOK/8, exact
  const int b    = tok / NS;
  const int s    = tok % NS;

  for (int i = lane; i < 118; i += 32) sx[w][i] = X[(size_t)tok * 118 + i];
  __syncthreads();

  // h[i][c] = ff(i,:) @ gat_W(:,c) + gat_b[c]
  for (int t = lane; t < 110; t += 32) {
    int i = t / 5, c = t % 5;
    float acc = gat_b[c];
    #pragma unroll
    for (int k = 0; k < 5; ++k) acc += sx[w][8 + i * 5 + k] * gat_W[k * 5 + c];
    sh[w][t] = acc;
  }
  __syncthreads();

  if (lane < 22) {
    float a1 = 0.f, a2 = 0.f;
    #pragma unroll
    for (int c = 0; c < 5; ++c) {
      float hv = sh[w][lane * 5 + c];
      a1 += hv * gat_a[c];
      a2 += hv * gat_a[5 + c];
    }
    ss1[w][lane] = a1;
    ss2[w][lane] = a2;
  }
  __syncthreads();

  // row-wise leaky-relu + softmax + aggregate -> hp (110, padded to 128)
  if (lane < 22) {
    const int i = lane;
    const float si = ss1[w][i];
    float m = -1e30f;
    for (int j = 0; j < 22; ++j) {
      float e = si + ss2[w][j];
      e = (e >= 0.f) ? e : 0.2f * e;
      m = fmaxf(m, e);
    }
    float sum = 0.f, hp[5] = {0.f, 0.f, 0.f, 0.f, 0.f};
    for (int j = 0; j < 22; ++j) {
      float e = si + ss2[w][j];
      e = (e >= 0.f) ? e : 0.2f * e;
      float p = expf(e - m);
      sum += p;
      #pragma unroll
      for (int c = 0; c < 5; ++c) hp[c] += p * sh[w][j * 5 + c];
    }
    float inv = 1.f / sum;
    #pragma unroll
    for (int c = 0; c < 5; ++c)
      hp16[(size_t)tok * 128 + i * 5 + c] = (_Float16)(hp[c] * inv);
  }
  if (lane < 18) hp16[(size_t)tok * 128 + 110 + lane] = (_Float16)0.f;  // K pad

  // embeddings + lin0 + positional encoding (first 48 features of src)
  float ang = (float)b * powf(100.0f, -2.0f * (float)s / (float)NS);
  float pe  = (s & 1) ? cosf(ang) : sinf(ang);
  if (lane < 16) {
    int act  = (int)sx[w][0];
    int zone = (int)sx[w][1];
    float od = lin0_b[lane];
    #pragma unroll
    for (int k = 0; k < 6; ++k) od += sx[w][2 + k] * lin0_W[k * 16 + lane];
    size_t base = (size_t)tok * 64;
    src32[base + lane]      = emb_act[act * 16 + lane]   + pe;
    src32[base + 16 + lane] = emb_zone[zone * 16 + lane] + pe;
    src32[base + 32 + lane] = od + pe;
  }
}

// ---------------------------------------------------------------------------
// WMMA GEMM: C(M x N) = relu?(A(M x K) @ W(K x N) + bias), f16 in / f32 acc /
// f16 out. One wave per 16x16 output tile. K multiple of 32. B pre-packed.
// A fragment (16-bit 16x32): lane L -> row M = L&15; e[0..7] = K 8*hi+0..7,
// e[8..15] = K 16+8*hi+0..7 (hi = L>=16)  => two 16B contiguous loads.
// ---------------------------------------------------------------------------
template <bool RELU>
__global__ __launch_bounds__(256) void gemm_wmma_f16(
    const _Float16* __restrict__ A, const _Float16* __restrict__ Bp,
    const float* __restrict__ bias, _Float16* __restrict__ out,
    int Mtiles, int Ntiles, int K, int N) {
  const int wave = threadIdx.x >> 5;
  const int lane = threadIdx.x & 31;
  const int tile = blockIdx.x * 8 + wave;
  if (tile >= Mtiles * Ntiles) return;          // wave-uniform: EXEC stays full
  const int mt = tile / Ntiles;
  const int nt = tile % Ntiles;
  const int hi  = lane >> 4;
  const int l15 = lane & 15;
  const int Ks  = K >> 5;

  const _Float16* arow  = A + (size_t)(mt * 16 + l15) * K + 8 * hi;
  const _Float16* bbase = Bp + ((size_t)(nt * Ks) * 32 + lane) * 16;

  v8f acc = {};
  for (int ks = 0; ks < Ks; ++ks) {
    union { v16h v; v8h h[2]; } a;
    a.h[0] = *(const v8h*)(arow + ks * 32);
    a.h[1] = *(const v8h*)(arow + ks * 32 + 16);
    v16h bfrag = *(const v16h*)(bbase + (size_t)ks * 512);
    acc = __builtin_amdgcn_wmma_f32_16x16x32_f16(
        /*neg_a=*/false, a.v, /*neg_b=*/false, bfrag,
        /*c_mod=*/(short)0, acc, /*reuse_a=*/false, /*reuse_b=*/false);
  }

  const int n = nt * 16 + l15;
  const float bn = bias[n];
  _Float16* orow = out + (size_t)(mt * 16 + 8 * hi) * N + n;
  #pragma unroll
  for (int v = 0; v < 8; ++v) {                 // C/D: row = 8*hi+v, col = l15
    float val = acc[v] + bn;
    if (RELU) val = val > 0.f ? val : 0.f;
    orow[(size_t)v * N] = (_Float16)val;
  }
}

// ---------------------------------------------------------------------------
// Stage A5: finish src assembly: src[:,48:64] = relu-MLP output + pe, and
// emit the f16 copy used by the qkv WMMA GEMM.
// ---------------------------------------------------------------------------
__global__ __launch_bounds__(256) void stage_a5(
    const _Float16* __restrict__ g2, float* __restrict__ src32,
    _Float16* __restrict__ src16) {
  const int tok = blockIdx.x * 256 + threadIdx.x;   // grid = NTOK/256, exact
  const int b = tok / NS, s = tok % NS;
  float ang = (float)b * powf(100.0f, -2.0f * (float)s / (float)NS);
  float pe  = (s & 1) ? cosf(ang) : sinf(ang);
  float* row = src32 + (size_t)tok * 64;
  #pragma unroll
  for (int j = 0; j < 16; ++j) row[48 + j] = (float)g2[(size_t)tok * 16 + j] + pe;
  _Float16* r16 = src16 + (size_t)tok * 64;
  #pragma unroll
  for (int d = 0; d < 64; ++d) r16[d] = (_Float16)row[d];
}

// ---------------------------------------------------------------------------
// Attention: only the LAST query position feeds the rest of the network, so
// each thread handles one (b, h): 40-key two-pass softmax over 8-dim head.
// ---------------------------------------------------------------------------
__global__ __launch_bounds__(256) void attn_kernel(
    const _Float16* __restrict__ qkv, float* __restrict__ attn_out) {
  const int idx = blockIdx.x * 256 + threadIdx.x;   // NB*NHEAD = 16384
  const int b = idx >> 3, h = idx & 7;
  const _Float16* qp = qkv + ((size_t)(b * NS + NS - 1)) * 192 + h * 8;
  float q[8];
  #pragma unroll
  for (int d = 0; d < 8; ++d) q[d] = (float)qp[d];
  const float scale = 0.35355339059327373f;         // 1/sqrt(8)
  float mx = -1e30f;
  for (int s = 0; s < NS; ++s) {
    const _Float16* kp = qkv + ((size_t)(b * NS + s)) * 192 + 64 + h * 8;
    float sc = 0.f;
    #pragma unroll
    for (int d = 0; d < 8; ++d) sc += q[d] * (float)kp[d];
    mx = fmaxf(mx, sc * scale);
  }
  float sum = 0.f, o[8] = {};
  for (int s = 0; s < NS; ++s) {
    const _Float16* kp = qkv + ((size_t)(b * NS + s)) * 192 + 64 + h * 8;
    const _Float16* vp = kp + 64;
    float sc = 0.f;
    #pragma unroll
    for (int d = 0; d < 8; ++d) sc += q[d] * (float)kp[d];
    float p = expf(sc * scale - mx);
    sum += p;
    #pragma unroll
    for (int d = 0; d < 8; ++d) o[d] += p * (float)vp[d];
  }
  float inv = 1.f / sum;
  #pragma unroll
  for (int d = 0; d < 8; ++d)
    attn_out[(size_t)b * 64 + h * 8 + d] = o[d] * inv;
}

// ---------------------------------------------------------------------------
// Tail: per batch row (2048 rows only): Wo + residual, LN1, FFN, LN2, head
// MLP chain, write 26 outputs. One 128-thread block per row.
// ---------------------------------------------------------------------------
struct TailW {
  const float *Wo, *bo, *ln1g, *ln1b, *W1, *b1, *W2, *b2, *ln2g, *ln2b;
  const float *reluW, *relub, *ndW, *ndb, *ldtW, *ldtb;
  const float *nzW, *nzb, *lzW, *lzb, *na0W, *na0b, *na1W, *na1b, *laW, *lab;
};

__global__ __launch_bounds__(128) void tail_kernel(
    const float* __restrict__ attn_out, const float* __restrict__ src32,
    float* __restrict__ out, TailW p) {
  const int b = blockIdx.x;
  const int t = threadIdx.x;
  __shared__ float sO[64], sX1[64], sX2[64], sH[256], sXr[64];
  __shared__ float sC[65], sZ[65], sFa[85], sA[85], sB[85];
  __shared__ float sMean, sVar, sdT;

  const float* orow = attn_out + (size_t)b * 64;
  const float* srow = src32 + ((size_t)(b * NS + NS - 1)) * 64;

  if (t < 64) {
    float acc = p.bo[t];
    for (int k = 0; k < 64; ++k) acc += orow[k] * p.Wo[k * 64 + t];
    sO[t] = acc + srow[t];
  }
  __syncthreads();
  if (t == 0) {
    float m = 0.f; for (int k = 0; k < 64; ++k) m += sO[k]; m *= (1.f / 64.f);
    float v = 0.f; for (int k = 0; k < 64; ++k) { float d = sO[k] - m; v += d * d; }
    sMean = m; sVar = v * (1.f / 64.f);
  }
  __syncthreads();
  if (t < 64) sX1[t] = (sO[t] - sMean) * rsqrtf(sVar + 1e-5f) * p.ln1g[t] + p.ln1b[t];
  __syncthreads();
  for (int j = t; j < 256; j += 128) {
    float acc = p.b1[j];
    for (int k = 0; k < 64; ++k) acc += sX1[k] * p.W1[k * 256 + j];
    sH[j] = acc > 0.f ? acc : 0.f;
  }
  __syncthreads();
  if (t < 64) {
    float acc = p.b2[t];
    for (int j = 0; j < 256; ++j) acc += sH[j] * p.W2[j * 64 + t];
    sO[t] = sX1[t] + acc;
  }
  __syncthreads();
  if (t == 0) {
    float m = 0.f; for (int k = 0; k < 64; ++k) m += sO[k]; m *= (1.f / 64.f);
    float v = 0.f; for (int k = 0; k < 64; ++k) { float d = sO[k] - m; v += d * d; }
    sMean = m; sVar = v * (1.f / 64.f);
  }
  __syncthreads();
  if (t < 64) sX2[t] = (sO[t] - sMean) * rsqrtf(sVar + 1e-5f) * p.ln2g[t] + p.ln2b[t];
  __syncthreads();
  if (t < 64) {
    float acc = p.relub[t];
    for (int k = 0; k < 64; ++k) acc += sX2[k] * p.reluW[k * 64 + t];
    sXr[t] = acc;
  }
  __syncthreads();
  if (t < 64) {
    float acc = p.ndb[t];
    for (int k = 0; k < 64; ++k) acc += sXr[k] * p.ndW[k * 64 + t];
    sO[t] = acc;                        // reuse sO as nd output
  }
  __syncthreads();
  if (t == 0) {
    float acc = p.ldtb[0];
    for (int k = 0; k < 64; ++k) acc += sO[k] * p.ldtW[k];
    sdT = acc;
  }
  __syncthreads();
  if (t == 0) sC[0] = sdT;
  if (t < 64) sC[1 + t] = sXr[t];
  __syncthreads();
  if (t < 65) {
    float acc = p.nzb[t];
    for (int k = 0; k < 65; ++k) acc += sC[k] * p.nzW[k * 65 + t];
    sZ[t] = acc;
  }
  __syncthreads();
  if (t < 20) {
    float acc = p.lzb[t];
    for (int k = 0; k < 65; ++k) acc += sZ[k] * p.lzW[k * 20 + t];
    sFa[t] = acc;                       // mz -> fa[0:20]
  }
  if (t == 0) sFa[20] = sdT;
  if (t < 64) sFa[21 + t] = sXr[t];
  __syncthreads();
  if (t < 85) {
    float acc = p.na0b[t];
    for (int k = 0; k < 85; ++k) acc += sFa[k] * p.na0W[k * 85 + t];
    sA[t] = acc;
  }
  __syncthreads();
  if (t < 85) {
    float acc = p.na1b[t];
    for (int k = 0; k < 85; ++k) acc += sA[k] * p.na1W[k * 85 + t];
    sB[t] = acc;
  }
  __syncthreads();
  float* orow_out = out + (size_t)b * 26;
  if (t == 0) orow_out[0] = sdT;
  if (t < 20) orow_out[1 + t] = sFa[t];
  if (t < 5) {
    float acc = p.lab[t];
    for (int k = 0; k < 85; ++k) acc += sB[k] * p.laW[k * 5 + t];
    orow_out[21 + t] = acc;
  }
}

// ---------------------------------------------------------------------------
// Host-side launch
// ---------------------------------------------------------------------------
extern "C" void kernel_launch(void* const* d_in, const int* in_sizes, int n_in,
                              void* d_out, int out_size, void* d_ws, size_t ws_size,
                              hipStream_t stream) {
  (void)in_sizes; (void)n_in; (void)out_size; (void)ws_size;
  const float* X        = (const float*)d_in[0];
  const float* emb_act  = (const float*)d_in[1];
  const float* emb_zone = (const float*)d_in[2];
  const float* lin0_W   = (const float*)d_in[3];
  const float* lin0_b   = (const float*)d_in[4];
  const float* gat_W    = (const float*)d_in[5];
  const float* gat_b    = (const float*)d_in[6];
  const float* gat_a    = (const float*)d_in[7];
  const float* gl0_W    = (const float*)d_in[8];
  const float* gl0_b    = (const float*)d_in[9];
  const float* gl1_W    = (const float*)d_in[10];
  const float* gl1_b    = (const float*)d_in[11];
  const float* gl2_W    = (const float*)d_in[12];
  const float* gl2_b    = (const float*)d_in[13];
  const float* Wqkv     = (const float*)d_in[14];
  const float* bqkv     = (const float*)d_in[15];

  char* ws = (char*)d_ws;
  float*    src32 = (float*)(ws + OFF_SRC32);
  _Float16* src16 = (_Float16*)(ws + OFF_SRC16);
  _Float16* pkGl0 = (_Float16*)(ws + OFF_PK_GL0);
  _Float16* pkGl1 = (_Float16*)(ws + OFF_PK_GL1);
  _Float16* pkGl2 = (_Float16*)(ws + OFF_PK_GL2);
  _Float16* pkQkv = (_Float16*)(ws + OFF_PK_QKV);
  _Float16* hp16  = (_Float16*)(ws + OFF_HP);
  _Float16* g0    = (_Float16*)(ws + OFF_G0);
  _Float16* g1    = (_Float16*)(ws + OFF_G1);
  _Float16* g2    = (_Float16*)(ws + OFF_G2);
  _Float16* qkv16 = (_Float16*)(ws + OFF_QKV);   // aliases hp16/g0 (dead by then)
  float*    attnO = (float*)(ws + OFF_ATTN);

  // 1) pack weights into WMMA B-fragment layout (Korig, N, Ksteps, Ntiles)
  pack_b<<<32, 256, 0, stream>>>(gl0_W, pkGl0, 110, 64, 4, 4);   // 110->128 pad
  pack_b<<<16, 256, 0, stream>>>(gl1_W, pkGl1, 64, 64, 2, 4);
  pack_b<<<4,  256, 0, stream>>>(gl2_W, pkGl2, 64, 16, 2, 1);
  pack_b<<<48, 256, 0, stream>>>(Wqkv,  pkQkv, 64, 192, 2, 12);

  // 2) per-token features -> hp (f16, K=128) + src[:,0:48]
  stage_a1<<<NTOK / 8, 256, 0, stream>>>(X, emb_act, emb_zone, lin0_W, lin0_b,
                                         gat_W, gat_b, gat_a, hp16, src32);

  // 3) WMMA MLP chain: hp(128)->64->64->16, relu epilogues
  const int Mt = NTOK / 16;   // 5120
  gemm_wmma_f16<true><<<(Mt * 4 + 7) / 8, 256, 0, stream>>>(hp16, pkGl0, gl0_b, g0, Mt, 4, 128, 64);
  gemm_wmma_f16<true><<<(Mt * 4 + 7) / 8, 256, 0, stream>>>(g0,   pkGl1, gl1_b, g1, Mt, 4, 64, 64);
  gemm_wmma_f16<true><<<(Mt * 1 + 7) / 8, 256, 0, stream>>>(g1,   pkGl2, gl2_b, g2, Mt, 1, 64, 16);

  // 4) assemble src (f32 + f16) with positional encoding
  stage_a5<<<NTOK / 256, 256, 0, stream>>>(g2, src32, src16);

  // 5) qkv = src @ Wqkv + bqkv (WMMA, f16 out)
  gemm_wmma_f16<false><<<(Mt * 12 + 7) / 8, 256, 0, stream>>>(src16, pkQkv, bqkv, qkv16, Mt, 12, 64, 192);

  // 6) attention at the last query position only
  attn_kernel<<<(NB * NHEAD) / 256, 256, 0, stream>>>(qkv16, attnO);

  // 7) tail: Wo + residual + LN1 + FFN + LN2 + head MLP (2048 rows)
  TailW p;
  p.Wo = (const float*)d_in[16];   p.bo = (const float*)d_in[17];
  p.ln1g = (const float*)d_in[18]; p.ln1b = (const float*)d_in[19];
  p.W1 = (const float*)d_in[20];   p.b1 = (const float*)d_in[21];
  p.W2 = (const float*)d_in[22];   p.b2 = (const float*)d_in[23];
  p.ln2g = (const float*)d_in[24]; p.ln2b = (const float*)d_in[25];
  p.reluW = (const float*)d_in[26]; p.relub = (const float*)d_in[27];
  p.ndW = (const float*)d_in[28];  p.ndb = (const float*)d_in[29];
  p.ldtW = (const float*)d_in[30]; p.ldtb = (const float*)d_in[31];
  p.nzW = (const float*)d_in[32];  p.nzb = (const float*)d_in[33];
  p.lzW = (const float*)d_in[34];  p.lzb = (const float*)d_in[35];
  p.na0W = (const float*)d_in[36]; p.na0b = (const float*)d_in[37];
  p.na1W = (const float*)d_in[38]; p.na1b = (const float*)d_in[39];
  p.laW = (const float*)d_in[40];  p.lab = (const float*)d_in[41];
  tail_kernel<<<NB, 128, 0, stream>>>(attnO, src32, (float*)d_out, p);
}
